// LocationBasedAttentionBlock_39994735460679
// MI455X (gfx1250) — matrix operand order repaired and checked by reading, not compiled, not run on hardware
//
#include <hip/hip_runtime.h>
#include <hip/hip_bf16.h>

// ---------------- problem constants ----------------
#define B_DIM 16
#define N_DIM 512
#define U_DIM 512
#define H_DIM 32
#define T_DIM 1024          // H*H
#define C_DIM 32            // T/H
#define G_DIM 512           // C*B
#define NG    8             // g-axis chunks for the scores pass
#define GCHUNK (G_DIM / NG) // 64
#define NEGV  (-4294967295.0f)   // -2^32 + 1
#define KEEP24 15099494u         // floor(0.9 * 2^24)
#define INV_KEEP (1.0f / 0.9f)

// ---------------- WMMA types / helpers ----------------
typedef __attribute__((ext_vector_type(16))) __bf16 v16bf;
typedef __attribute__((ext_vector_type(8)))  __bf16 v8bf;
typedef __attribute__((ext_vector_type(8)))  float  v8f;

__device__ __forceinline__ v8f wmma_bf16(v16bf a, v16bf b, v8f c) {
    // D = A(16x32 bf16) * B(32x16 bf16) + C(16x16 f32)
    return __builtin_amdgcn_wmma_f32_16x16x32_bf16(
        false, a, false, b, (short)0, c, false, false);
}

// A fragment: 16x32, row-major LDS tile (K contiguous), stride in bf16 elems
__device__ __forceinline__ v16bf frag_A(const __bf16* base, int rowbase, int stride) {
    const int lane = threadIdx.x & 31;
    const int half = lane >> 4;
    const int r    = lane & 15;
    const __bf16* p = base + (rowbase + r) * stride;
    v8bf lo = *(const v8bf*)(p + 8 * half);
    v8bf hi = *(const v8bf*)(p + 16 + 8 * half);
    v16bf o;
#pragma unroll
    for (int e = 0; e < 8; ++e) { o[e] = lo[e]; o[e + 8] = hi[e]; }
    return o;
}

// B fragment: 32x16, LDS tile stored column-major as [col][k] with k contiguous
__device__ __forceinline__ v16bf frag_B(const __bf16* base, int colbase, int stride) {
    const int lane = threadIdx.x & 31;
    const int half = lane >> 4;
    const int r    = lane & 15;
    const __bf16* p = base + (colbase + r) * stride;
    v8bf lo = *(const v8bf*)(p + 8 * half);
    v8bf hi = *(const v8bf*)(p + 16 + 8 * half);
    v16bf o;
#pragma unroll
    for (int e = 0; e < 8; ++e) { o[e] = lo[e]; o[e + 8] = hi[e]; }
    return o;
}

__device__ __forceinline__ unsigned hash_u32(unsigned x) {
    x ^= x >> 16; x *= 0x7FEB352Du;
    x ^= x >> 15; x *= 0x846CA68Bu;
    x ^= x >> 16; return x;
}

// CDNA5 async copy: 16B global -> LDS, tracked by ASYNCcnt (in-order done)
__device__ __forceinline__ void async_copy_b128(unsigned lds_off, const void* gptr) {
    unsigned long long ga = (unsigned long long)gptr;
    asm volatile("global_load_async_to_lds_b128 %0, %1, off"
                 :: "v"(lds_off), "v"(ga) : "memory");
}
__device__ __forceinline__ void wait_async0() {
    asm volatile("s_wait_asynccnt 0x0" ::: "memory");
}
__device__ __forceinline__ void wait_async1() {
    asm volatile("s_wait_asynccnt 0x1" ::: "memory");
}

// =====================================================================
// Kernel 1: projections  Y = W @ x[b]  -> heads-layout bf16 [g][n][h]
// grid (T/64, N/128, 3*B), block 256 (8 waves); tile 128n x 64t, wave = 32x32
// =====================================================================
__global__ void proj_kernel(const float* __restrict__ x,
                            const float* __restrict__ Qp,
                            const float* __restrict__ Kp,
                            const float* __restrict__ Vp,
                            __bf16* __restrict__ Qh,
                            __bf16* __restrict__ Kh,
                            __bf16* __restrict__ Vh) {
    __shared__ __bf16 sA[128 * 40];   // W tile  [n][k], k contiguous
    __shared__ __bf16 sB[64 * 40];    // x tile transposed [t][k], k contiguous

    const int tid  = threadIdx.x;
    const int widx = blockIdx.z / B_DIM;
    const int b    = blockIdx.z % B_DIM;
    const float*  W   = (widx == 0) ? Qp : (widx == 1) ? Kp : Vp;
    __bf16*       dst = (widx == 0) ? Qh : (widx == 1) ? Kh : Vh;

    const int nbase = blockIdx.y * 128;
    const int t0    = blockIdx.x * 64;

    const int w  = tid >> 5;       // wave id 0..7
    const int wn = w >> 1;         // 0..3
    const int wt = w & 1;          // 0..1
    const int ra = wn * 32;
    const int cb = wt * 32;

    v8f acc[2][2] = {};

    for (int k0 = 0; k0 < U_DIM; k0 += 32) {
        for (int idx = tid; idx < 128 * 32; idx += 256) {
            int i = idx >> 5, kk = idx & 31;
            sA[i * 40 + kk] = (__bf16)W[(nbase + i) * U_DIM + k0 + kk];
        }
        for (int idx = tid; idx < 64 * 32; idx += 256) {
            int uu = idx >> 6, tt = idx & 63;
            sB[tt * 40 + uu] = (__bf16)x[((size_t)b * U_DIM + k0 + uu) * T_DIM + t0 + tt];
        }
        __syncthreads();

        v16bf a0 = frag_A(sA, ra,      40);
        v16bf a1 = frag_A(sA, ra + 16, 40);
        v16bf b0 = frag_B(sB, cb,      40);
        v16bf b1 = frag_B(sB, cb + 16, 40);
        acc[0][0] = wmma_bf16(a0, b0, acc[0][0]);
        acc[0][1] = wmma_bf16(a0, b1, acc[0][1]);
        acc[1][0] = wmma_bf16(a1, b0, acc[1][0]);
        acc[1][1] = wmma_bf16(a1, b1, acc[1][1]);
        __syncthreads();
    }

    const int lane = tid & 31;
    const int half = lane >> 4;
    const int r    = lane & 15;
#pragma unroll
    for (int i = 0; i < 2; ++i)
#pragma unroll
        for (int j = 0; j < 2; ++j)
#pragma unroll
            for (int v = 0; v < 8; ++v) {
                int n = nbase + ra + i * 16 + 8 * half + v;
                int t = t0 + cb + j * 16 + r;
                int c = t >> 5, h = t & 31;
                int g = c * B_DIM + b;
                dst[((size_t)g * N_DIM + n) * H_DIM + h] = (__bf16)acc[i][j][v];
            }
}

// =====================================================================
// Kernel 2: scores + partial online softmax stats over a g-chunk
// grid (N/64, N/64, NG), block 512 (16 waves); wave = one 16x16 (n,m) tile
// Qh/Kh tiles: double-buffered global_load_async_to_lds_b128 (ASYNCcnt),
// one copy always in flight behind the WMMA + exp work.
// =====================================================================
__global__ void scores_kernel(const __bf16* __restrict__ Qh,
                              const __bf16* __restrict__ Kh,
                              const int*    __restrict__ pm,
                              float* __restrict__ scores,
                              float* __restrict__ Mp,
                              float* __restrict__ Zp) {
    __shared__ __bf16 sQ[2][64 * 40];    // [n][h], rows padded to 80B
    __shared__ __bf16 sK[2][64 * 40];    // [m][h]

    const int tid = threadIdx.x;
    const int nb = blockIdx.y * 64;
    const int mb = blockIdx.x * 64;
    const int g0 = blockIdx.z * GCHUNK;

    const int w  = tid >> 5;
    const int wn = w >> 2;     // 0..3
    const int wm = w & 3;      // 0..3
    const int lane = tid & 31;
    const int half = lane >> 4;
    const int r    = lane & 15;
    const int mg   = mb + wm * 16 + r;

    // async staging assignment: thread -> one 16B chunk of one tile
    const int tilesel = tid >> 8;          // 0 -> sQ, 1 -> sK
    const int cid     = tid & 255;
    const int nn      = cid >> 2;          // row 0..63
    const int kk      = cid & 3;           // 16B chunk in 64B row
    const unsigned chunk = nn * 80 + kk * 16;
    const unsigned lds0 =
        (unsigned)(size_t)(tilesel ? &sK[0][0] : &sQ[0][0]) + chunk;
    const unsigned lds1 =
        (unsigned)(size_t)(tilesel ? &sK[1][0] : &sQ[1][0]) + chunk;
    const __bf16* srcrow = (tilesel ? Kh : Qh) +
        ((size_t)g0 * N_DIM + (tilesel ? mb : nb) + nn) * H_DIM;
    const char* gaddr = (const char*)srcrow + kk * 16;
    const size_t gstep = (size_t)N_DIM * H_DIM * sizeof(__bf16);  // per-g stride

    float runM[8], runZ[8];
#pragma unroll
    for (int v = 0; v < 8; ++v) { runM[v] = -3.0e38f; runZ[v] = 0.0f; }

    // prime the pipeline: copy for gi=0 into buffer 0
    async_copy_b128(lds0, gaddr);
    gaddr += gstep;

    for (int gi = 0; gi < GCHUNK; ++gi) {
        const int g = g0 + gi;
        if (gi + 1 < GCHUNK) {
            // issue next copy into the other buffer, keep 1 in flight
            async_copy_b128((gi & 1) ? lds0 : lds1, gaddr);
            gaddr += gstep;
            wait_async1();        // current buffer's copy has landed (in-order)
        } else {
            wait_async0();
        }
        __syncthreads();

        const __bf16* q = sQ[gi & 1];
        const __bf16* k = sK[gi & 1];
        v16bf a  = frag_A(q, wn * 16, 40);
        v16bf bq = frag_B(k, wm * 16, 40);
        v8f z = {};
        v8f acc = wmma_bf16(a, bq, z);

        const bool masked = (pm[(g & (B_DIM - 1)) * N_DIM + mg] == 0);
#pragma unroll
        for (int v = 0; v < 8; ++v) {
            int n = nb + wn * 16 + 8 * half + v;
            float s = masked ? NEGV : acc[v];
            scores[((size_t)g * N_DIM + n) * N_DIM + mg] = s;
            float nm = fmaxf(runM[v], s);
            runZ[v] = runZ[v] * __expf(runM[v] - nm) + __expf(s - nm);
            runM[v] = nm;
        }
        __syncthreads();   // all waves done reading buf[gi&1] before reuse
    }

#pragma unroll
    for (int v = 0; v < 8; ++v) {
        int n = nb + wn * 16 + 8 * half + v;
        size_t o = (size_t)blockIdx.z * N_DIM * N_DIM + n * N_DIM + mg;
        Mp[o] = runM[v];
        Zp[o] = runZ[v];
    }
}

// =====================================================================
// Kernel 2b: merge NG partial (M,Z) stats -> fused log-sum-exp
//            lse[n,m] = M + log(Z)   (single f32 per element for pass 3)
// =====================================================================
__global__ void stats_combine(const float* __restrict__ Mp,
                              const float* __restrict__ Zp,
                              float* __restrict__ Lse) {
    const int i = blockIdx.x * 256 + threadIdx.x;   // < N*N
    float m = -3.0e38f;
#pragma unroll
    for (int c = 0; c < NG; ++c)
        m = fmaxf(m, Mp[(size_t)c * N_DIM * N_DIM + i]);
    float z = 0.0f;
#pragma unroll
    for (int c = 0; c < NG; ++c)
        z += Zp[(size_t)c * N_DIM * N_DIM + i] *
             __expf(Mp[(size_t)c * N_DIM * N_DIM + i] - m);
    Lse[i] = m + __logf(z);
}

// =====================================================================
// Kernel 3: probs = exp(s - lse) + dropout -> ctx GEMM -> permuted output
// grid (N/64, G), block 256 (8 waves); tile 64n x 32h per workgroup
// =====================================================================
__global__ void ctx_kernel(const __bf16* __restrict__ Vh,
                           const float*  __restrict__ scores,
                           const float*  __restrict__ Lse,
                           float* __restrict__ out,
                           float* __restrict__ att) {
    __shared__ __bf16 sP[64 * 40];    // probs tile [n][m] (A frags)
    __shared__ __bf16 sV[32 * 40];    // Vh tile transposed [h][m] (B frags)

    const int tid = threadIdx.x;
    const int nb  = blockIdx.x * 64;
    const int g   = blockIdx.y;

    const int w  = tid >> 5;
    const int wn = w >> 1;     // 0..3
    const int wh = w & 1;      // 0..1
    const int lane = tid & 31;
    const int half = lane >> 4;
    const int r    = lane & 15;

    v8f acc = {};

    for (int m0 = 0; m0 < N_DIM; m0 += 32) {
        for (int idx = tid; idx < 64 * 32; idx += 256) {
            int nn = idx >> 5, mm = idx & 31;
            int n = nb + nn, m = m0 + mm;
            int si = (g * N_DIM + n) * N_DIM + m;     // < 2^27, fits int
            if (mm == 0 && m0 + 32 < N_DIM)
                __builtin_prefetch(&scores[si + 32], 0, 1);   // global_prefetch_b8
            float p = __expf(scores[si] - Lse[n * N_DIM + m]);
            unsigned hh = hash_u32(0x9E3779B9u ^ (unsigned)si);
            p = ((hh & 0xFFFFFFu) < KEEP24) ? p * INV_KEEP : 0.0f;
            sP[nn * 40 + mm] = (__bf16)p;
            if (n == N_DIM - 1) att[g * N_DIM + m] = p;   // attention side output
        }
        for (int idx = tid; idx < 32 * 32; idx += 256) {
            int mm = idx >> 5, hv = idx & 31;
            sV[hv * 40 + mm] = Vh[((size_t)g * N_DIM + m0 + mm) * H_DIM + hv];
        }
        __syncthreads();

        v16bf a  = frag_A(sP, wn * 16, 40);
        v16bf bv = frag_B(sV, wh * 16, 40);
        acc = wmma_bf16(a, bv, acc);
        __syncthreads();
    }

    const int a0 = g >> 5, a1 = g & 31;
#pragma unroll
    for (int v = 0; v < 8; ++v) {
        int n = nb + wn * 16 + 8 * half + v;
        int h = wh * 16 + r;
        int oi = (a1 * N_DIM + n) * (B_DIM * H_DIM) + a0 * H_DIM + h;
        unsigned hh = hash_u32(0x85EBCA6Bu ^ (unsigned)oi);
        float val = ((hh & 0xFFFFFFu) < KEEP24) ? acc[v] * INV_KEEP : 0.0f;
        out[oi] = val;
    }
}

// =====================================================================
// launch
// =====================================================================
extern "C" void kernel_launch(void* const* d_in, const int* in_sizes, int n_in,
                              void* d_out, int out_size, void* d_ws, size_t ws_size,
                              hipStream_t stream) {
    const float* x  = (const float*)d_in[0];
    const float* Qp = (const float*)d_in[1];   // [1,N,U] == [N,U]
    const float* Kp = (const float*)d_in[2];
    const float* Vp = (const float*)d_in[3];
    const int*   pm = (const int*)d_in[4];

    char* ws = (char*)d_ws;
    const size_t HEADS_BYTES = (size_t)G_DIM * N_DIM * H_DIM * sizeof(__bf16); // 16 MB
    const size_t NN_BYTES    = (size_t)N_DIM * N_DIM * sizeof(float);          //  1 MB
    __bf16* Qh = (__bf16*)(ws);
    __bf16* Kh = (__bf16*)(ws + HEADS_BYTES);
    __bf16* Vh = (__bf16*)(ws + 2 * HEADS_BYTES);
    float*  Lse = (float*)(ws + 3 * HEADS_BYTES);
    float*  Mp  = (float*)(ws + 3 * HEADS_BYTES + NN_BYTES);
    float*  Zp  = (float*)(ws + 3 * HEADS_BYTES + (1 + NG) * NN_BYTES);
    float*  scores = (float*)(ws + 3 * HEADS_BYTES + (1 + 2 * NG) * NN_BYTES);

    float* out = (float*)d_out;                                 // [32,512,512]
    float* att = out + (size_t)H_DIM * N_DIM * B_DIM * H_DIM;   // [16,32,512]

    dim3 g1(T_DIM / 64, N_DIM / 128, 3 * B_DIM);
    proj_kernel<<<g1, dim3(256), 0, stream>>>(x, Qp, Kp, Vp, Qh, Kh, Vh);

    dim3 g2(N_DIM / 64, N_DIM / 64, NG);
    scores_kernel<<<g2, dim3(512), 0, stream>>>(Qh, Kh, pm, scores, Mp, Zp);

    stats_combine<<<dim3((N_DIM * N_DIM) / 256), dim3(256), 0, stream>>>(Mp, Zp, Lse);

    dim3 g3(N_DIM / 64, G_DIM, 1);
    ctx_kernel<<<g3, dim3(256), 0, stream>>>(Vh, scores, Lse, out, att);
}
